// SelectedUnitsHead_54803782697196
// MI455X (gfx1250) — compile-verified, matrix operand order checked by reading, not run on hardware
//
#include <hip/hip_runtime.h>
#include <hip/hip_bf16.h>
#include <math.h>

// ---------------- sizes ----------------
constexpr int kB   = 256;   // batch
constexpr int kE   = 512;   // entities
constexpr int kD   = 256;   // entity embedding
constexpr int kK   = 32;    // K32
constexpr int kH   = 256;   // H256
constexpr int kAE  = 1024;  // autoregressive embedding
constexpr int kMUT = 256;   // max unit type
constexpr int kSEL = 64;    // MAXSEL
constexpr float kNEG = -1e9f;
constexpr float kTINY = 1.17549435e-38f;

typedef __bf16 bf16_t;
typedef __attribute__((ext_vector_type(16))) __bf16 v16bf;
typedef __attribute__((ext_vector_type(8)))  __bf16 v8bf;
typedef __attribute__((ext_vector_type(8)))  float  v8f;

// ---------------- WMMA tile loaders (16x32 bf16 A-layout, wave32) ----------------
// 16-bit A-matrix 16x32 layout (ISA 7.12.2): lane l<16 row M=l holds K {k0..k0+7}
// in elems 0..7 and K {k0+16..k0+23} in elems 8..15; lanes 16..31 are the +8 K shift.
__device__ __forceinline__ v16bf load_tile_bf16(const bf16_t* base, int ld, int row0,
                                                int k0, int lane) {
  int l = lane & 15, hi = lane >> 4;
  const bf16_t* p = base + (size_t)(row0 + l) * ld + k0 + hi * 8;
  v8bf p0 = *(const v8bf*)p;
  v8bf p1 = *(const v8bf*)(p + 16);
  v16bf a;
#pragma unroll
  for (int j = 0; j < 8; ++j) { a[j] = p0[j]; a[8 + j] = p1[j]; }
  return a;
}

__device__ __forceinline__ v16bf load_tile_f32src(const float* base, int ld, int row0,
                                                  int k0, int lane) {
  int l = lane & 15, hi = lane >> 4;
  const float* p = base + (size_t)(row0 + l) * ld + k0 + hi * 8;
  v16bf a;
#pragma unroll
  for (int j = 0; j < 8; ++j) { a[j] = (bf16_t)p[j]; a[8 + j] = (bf16_t)p[j + 16]; }
  return a;
}

// ---------------- threefry2x32 (JAX-compatible) ----------------
__device__ __forceinline__ unsigned rotl32(unsigned x, int r) {
  return (x << r) | (x >> (32 - r));
}
__device__ __forceinline__ void threefry2x32(unsigned k0, unsigned k1,
                                             unsigned c0, unsigned c1,
                                             unsigned& o0, unsigned& o1) {
  unsigned ks2 = 0x1BD11BDAu ^ k0 ^ k1;
  unsigned x0 = c0 + k0, x1 = c1 + k1;
#pragma unroll
  for (int j = 0; j < 4; ++j) { const int R[4] = {13,15,26,6};
    x0 += x1; x1 = rotl32(x1, R[j]); x1 ^= x0; }
  x0 += k1; x1 += ks2 + 1u;
#pragma unroll
  for (int j = 0; j < 4; ++j) { const int R[4] = {17,29,16,24};
    x0 += x1; x1 = rotl32(x1, R[j]); x1 ^= x0; }
  x0 += ks2; x1 += k0 + 2u;
#pragma unroll
  for (int j = 0; j < 4; ++j) { const int R[4] = {13,15,26,6};
    x0 += x1; x1 = rotl32(x1, R[j]); x1 ^= x0; }
  x0 += k0; x1 += k1 + 3u;
#pragma unroll
  for (int j = 0; j < 4; ++j) { const int R[4] = {17,29,16,24};
    x0 += x1; x1 = rotl32(x1, R[j]); x1 ^= x0; }
  x0 += k1; x1 += ks2 + 4u;
  o0 = x0; o1 = x1;
}

__device__ __forceinline__ float sigmoidf(float x) { return 1.0f / (1.0f + expf(-x)); }

// ---------------- kernel: convert weights to bf16 ----------------
__global__ void k_convert(const float* fc1, const float* fw, const float* cw,
                          bf16_t* fc1b, bf16_t* fwb, bf16_t* cwb) {
  int i = blockIdx.x * blockDim.x + threadIdx.x;
  if (i < kH * kAE)  fc1b[i] = (bf16_t)fc1[i];
  if (i < kH * kMUT) fwb[i]  = (bf16_t)fw[i];
  if (i < kK * kD)   cwb[i]  = (bf16_t)cw[i];
}

// ---------------- kernel: the_func_embed = relu(uoh @ funcW^T + b)  [B,256] ----------------
__global__ void k_func_gemm(const float* uoh, const bf16_t* fwb, const float* bias,
                            float* tfe) {
  int lane = threadIdx.x & 31;
  int tile = blockIdx.x * 8 + (threadIdx.x >> 5);   // 256 tiles of 16x16
  int row0 = (tile >> 4) * 16, col0 = (tile & 15) * 16;
  v8f acc = {};
#pragma unroll
  for (int k0 = 0; k0 < kMUT; k0 += 32) {
    v16bf a = load_tile_f32src(uoh, kMUT, row0, k0, lane);
    v16bf b = load_tile_bf16(fwb, kMUT, col0, k0, lane);
    acc = __builtin_amdgcn_wmma_f32_16x16x32_bf16(false, a, false, b, (short)0, acc,
                                                  false, false);
  }
  int l = lane & 15, hi = lane >> 4;
  float bv = bias[col0 + l];
#pragma unroll
  for (int r = 0; r < 8; ++r)
    tfe[(size_t)(row0 + r + hi * 8) * kH + col0 + l] = fmaxf(acc[r] + bv, 0.0f);
}

// ---------------- kernel: key_t = ent @ convW^T + b   [B*E, 32] ----------------
// One wave per 16-row tile computes BOTH 16-wide column halves: the 134 MB entity
// stream is loaded exactly once, two WMMAs per A-tile load.
__global__ void k_keyt_gemm(const float* ent, const bf16_t* cwb, const float* cb,
                            float* keyt) {
  int lane = threadIdx.x & 31;
  int rtile = blockIdx.x * 8 + (threadIdx.x >> 5);  // 8192 row tiles
  int row0 = rtile * 16;
  v8f acc0 = {}, acc1 = {};
#pragma unroll
  for (int k0 = 0; k0 < kD; k0 += 32) {
    v16bf a  = load_tile_f32src(ent, kD, row0, k0, lane);
    v16bf b0 = load_tile_bf16(cwb, kD, 0,  k0, lane);
    v16bf b1 = load_tile_bf16(cwb, kD, 16, k0, lane);
    acc0 = __builtin_amdgcn_wmma_f32_16x16x32_bf16(false, a, false, b0, (short)0, acc0,
                                                   false, false);
    acc1 = __builtin_amdgcn_wmma_f32_16x16x32_bf16(false, a, false, b1, (short)0, acc1,
                                                   false, false);
  }
  int l = lane & 15, hi = lane >> 4;
  float bv0 = cb[l], bv1 = cb[16 + l];
#pragma unroll
  for (int r = 0; r < 8; ++r) {
    size_t row = (size_t)(row0 + r + hi * 8);
    keyt[row * kK + l]      = acc0[r] + bv0;
    keyt[row * kK + 16 + l] = acc1[r] + bv1;
  }
}

// ---------------- kernel: per-batch init ----------------
__global__ void k_batch_init(float* keyt, const int* enumv, const float* newv,
                             const float* ae_in, float* keyavg, unsigned* maskb,
                             float* h, float* c, float* ae, bf16_t* aeb,
                             unsigned* ended, int* sel, int* gcnt) {
  int b = blockIdx.x, t = threadIdx.x;
  int en = enumv[b];
  float* kt = keyt + (size_t)b * kE * kK;
  if (t < kK) {
    kt[(kE - 1) * kK + t] = 0.0f;      // zero last slot
    kt[en * kK + t] = newv[t];          // end-token embedding at end index
  }
  __syncthreads();
  if (t < kK) {
    float s = 0.0f;
    for (int e = 0; e <= en; ++e) s += kt[e * kK + t];
    keyavg[b * kK + t] = s / (float)en;
    h[b * kK + t] = 0.0f; c[b * kK + t] = 0.0f;
  }
  if (t < 16) {
    unsigned m = 0;
#pragma unroll
    for (int j = 0; j < 32; ++j) { int e = t * 32 + j; if (e < en) m |= (1u << j); }
    maskb[b * 16 + t] = m;              // selectable: e < enum (end token off at step 0)
  }
  for (int j = t; j < kAE; j += blockDim.x) {
    float v = ae_in[(size_t)b * kAE + j];
    ae[(size_t)b * kAE + j]  = v;
    aeb[(size_t)b * kAE + j] = (bf16_t)v;
  }
  if (t == 0) { ended[b] = 0u; sel[b] = kSEL; }
  if (b == 0 && t < kSEL) gcnt[t] = 0;
}

// ---------------- kernel: x = ae @ fc1W^T + b   [B,256]  (per step) ----------------
// One block per 16-row stripe of the batch. The 16x1024 bf16 A-stripe (32 KB,
// contiguous in memory) is staged into LDS once with gfx1250 async-to-LDS copies,
// then all 8 waves read it back via ds_load for their two 16-col output tiles.
__global__ void k_fc1_gemm(const bf16_t* aeb, const bf16_t* w, const float* bias,
                           float* xout) {
  __shared__ __align__(16) bf16_t sA[16 * kAE];   // 32 KB
  int t = threadIdx.x;
  int lane = t & 31, wv = t >> 5;
  int row0 = blockIdx.x * 16;                     // 16 blocks cover B=256 rows

  // async copy: 32 KB contiguous stripe, 256 threads x 16 B x 8 iters
  const bf16_t* gsrc = aeb + (size_t)row0 * kAE;
  unsigned lds_base = (unsigned)(size_t)(&sA[0]);   // low 32 bits == LDS address
#pragma unroll
  for (int it = 0; it < 8; ++it) {
    unsigned off = (unsigned)((it * 256 + t) * 16);
    unsigned laddr = lds_base + off;
    asm volatile("global_load_async_to_lds_b128 %0, %1, %2"
                 :: "v"(laddr), "v"(off), "s"(gsrc) : "memory");
  }
  asm volatile("s_wait_asynccnt 0x0" ::: "memory");
  __syncthreads();

  int l = lane & 15, hi = lane >> 4;
  int col0 = (wv * 2) * 16, col1 = col0 + 16;     // 2 output tiles per wave
  v8f acc0 = {}, acc1 = {};
#pragma unroll 4
  for (int k0 = 0; k0 < kAE; k0 += 32) {
    v16bf a;
    {
      int base = l * kAE + k0 + hi * 8;
#pragma unroll
      for (int j = 0; j < 8; ++j) { a[j] = sA[base + j]; a[8 + j] = sA[base + 16 + j]; }
    }
    v16bf b0 = load_tile_bf16(w, kAE, col0, k0, lane);
    v16bf b1 = load_tile_bf16(w, kAE, col1, k0, lane);
    acc0 = __builtin_amdgcn_wmma_f32_16x16x32_bf16(false, a, false, b0, (short)0, acc0,
                                                   false, false);
    acc1 = __builtin_amdgcn_wmma_f32_16x16x32_bf16(false, a, false, b1, (short)0, acc1,
                                                   false, false);
  }
  float bv0 = bias[col0 + l], bv1 = bias[col1 + l];
#pragma unroll
  for (int r = 0; r < 8; ++r) {
    size_t row = (size_t)(row0 + r + hi * 8);
    xout[row * kH + col0 + l] = acc0[r] + bv0;
    xout[row * kH + col1 + l] = acc1[r] + bv1;
  }
}

// ---------------- kernel: one autoregressive step (one block per batch) ----------------
__global__ void k_step(int i, const float* x, const float* tfe,
                       const float* fc2W, const float* fc2b,
                       const float* Wih, const float* Whh,
                       const float* bih, const float* bhh,
                       const float* projW, const float* projb,
                       const float* keyt, const float* keyavg,
                       const int* enumv, const unsigned char* selm,
                       unsigned* maskb, float* h, float* c,
                       float* ae, bf16_t* aeb,
                       unsigned* ended, int* sel, int* gcnt,
                       float* out_logits, float* out_units) {
  __shared__ float s_relu[kH];
  __shared__ float s_xk[kK], s_z[4 * kK], s_h[kK], s_hn[kK], s_cn[kK], s_o[kK];
  __shared__ float s_val[256];
  __shared__ int   s_idx[256];

  int b = blockIdx.x, t = threadIdx.x;
  int en = enumv[b];
  bool done = (i > 0) && (gcnt[i - 1] == kB);
  bool pad = done || (selm[b] == 0);
  float* lrow = out_logits + ((size_t)b * kSEL + i) * kE;

  if (done) {                                    // torch early-break emulation
    for (int e = t; e < kE; e += 256) lrow[e] = kNEG;
    if (t == 0) {
      out_units[b * kSEL + i] = (float)(kE - 1);
      atomicAdd(&gcnt[i], (int)ended[b]);        // keep the all-ended count = B
    }
    return;
  }
  if (t == 0 && i == 1)                          // end token becomes selectable
    maskb[b * 16 + (en >> 5)] |= (1u << (en & 31));

  // phase A: relu(x + the_func_embed)
  s_relu[t] = fmaxf(x[(size_t)b * kH + t] + tfe[(size_t)b * kH + t], 0.0f);
  if (t < kK) s_h[t] = h[b * kK + t];
  __syncthreads();

  // phase B: xk = relu(...) @ fc2W^T + b
  if (t < kK) {
    float s = fc2b[t];
    const float* wr = fc2W + t * kH;
    for (int j = 0; j < kH; ++j) s += s_relu[j] * wr[j];
    s_xk[t] = s;
  }
  __syncthreads();

  // phase C: LSTM cell (gate order i,f,g,o)
  if (t < 4 * kK) {
    float s = bih[t] + bhh[t];
    const float* wi = Wih + t * kK;
    const float* wh = Whh + t * kK;
#pragma unroll
    for (int k = 0; k < kK; ++k) s += s_xk[k] * wi[k] + s_h[k] * wh[k];
    s_z[t] = s;
  }
  __syncthreads();
  if (t < kK) {
    float zi = s_z[t], zf = s_z[kK + t], zg = s_z[2 * kK + t], zo = s_z[3 * kK + t];
    float cn = sigmoidf(zf) * c[b * kK + t] + sigmoidf(zi) * tanhf(zg);
    s_cn[t] = cn;
    s_hn[t] = sigmoidf(zo) * tanhf(cn);
  }
  __syncthreads();

  // step key = fold_in(key(42), i)
  unsigned sk0, sk1;
  threefry2x32(0u, 42u, 0u, (unsigned)i, sk0, sk1);

  // phase D: y = h_new . key_t, mask, Gumbel-argmax sampling
  float best = -3.0e38f; int bidx = 0;
#pragma unroll
  for (int rep = 0; rep < 2; ++rep) {
    int e = t + rep * 256;
    const float* kr = keyt + ((size_t)b * kE + e) * kK;
    if (rep == 0) __builtin_prefetch(keyt + ((size_t)b * kE + e + 256) * kK, 0, 0);
    float y = 0.0f;
#pragma unroll
    for (int k = 0; k < kK; ++k) y += s_hn[k] * kr[k];
    bool selb = (maskb[b * 16 + (e >> 5)] >> (e & 31)) & 1u;
    float logit = selb ? y : kNEG;
    lrow[e] = pad ? kNEG : logit;
    // JAX uniform bits: counts split in halves, word0 for j<n/2 else word1
    unsigned j = (unsigned)(b * kE + e);
    unsigned o0, o1, bits;
    if (j < 65536u) { threefry2x32(sk0, sk1, j, j + 65536u, o0, o1); bits = o0; }
    else            { threefry2x32(sk0, sk1, j - 65536u, j, o0, o1); bits = o1; }
    float u = __uint_as_float((bits >> 9) | 0x3f800000u) - 1.0f;
    u = u * (1.0f - kTINY) + kTINY;
    float g = -logf(-logf(u));
    float v = logit + g;
    if (v > best || (v == best && e < bidx)) { best = v; bidx = e; }
  }
  s_val[t] = best; s_idx[t] = bidx;
  __syncthreads();
  for (int s = 128; s > 0; s >>= 1) {
    if (t < s) {
      float v2 = s_val[t + s]; int i2 = s_idx[t + s];
      if (v2 > s_val[t] || (v2 == s_val[t] && i2 < s_idx[t])) {
        s_val[t] = v2; s_idx[t] = i2;
      }
    }
    __syncthreads();
  }
  int eid = s_idx[0];
  unsigned was = ended[b];
  __syncthreads();                               // all reads before state writes

  // phase E: state + output updates
  bool last = (eid == en);
  unsigned isend2 = was | (last ? 1u : 0u);
  if (t == 0) {
    maskb[b * 16 + (eid >> 5)] &= ~(1u << (eid & 31));
    if (last) sel[b] = i;
    ended[b] = isend2;
    out_units[b * kSEL + i] = pad ? (float)(kE - 1) : (float)eid;
    atomicAdd(&gcnt[i], (int)isend2);
  }
  if (t < kK) {
    s_o[t] = keyt[((size_t)b * kE + eid) * kK + t] - keyavg[b * kK + t];
    h[b * kK + t] = s_hn[t];
    c[b * kK + t] = s_cn[t];
  }
  __syncthreads();
  float scale = isend2 ? 0.0f : 1.0f;
#pragma unroll
  for (int r = 0; r < 4; ++r) {
    int idx = t + r * 256;                       // 0..1023
    float s = projb[idx];
    const float* pw = projW + idx * kK;
#pragma unroll
    for (int k = 0; k < kK; ++k) s += s_o[k] * pw[k];
    float nv = ae[(size_t)b * kAE + idx] + s * scale;
    ae[(size_t)b * kAE + idx]  = nv;
    aeb[(size_t)b * kAE + idx] = (bf16_t)nv;
  }
}

// ---------------- kernel: finalize (select_unit_mask override) ----------------
__global__ void k_finalize(const float* ae, const float* ae_in, const unsigned char* selm,
                           const int* sel, float* out_ae, float* out_sel) {
  int b = blockIdx.x, t = threadIdx.x;
  bool ns = (selm[b] == 0);
  for (int r = t; r < kAE; r += 256)
    out_ae[(size_t)b * kAE + r] = ns ? ae_in[(size_t)b * kAE + r]
                                     : ae[(size_t)b * kAE + r];
  if (t == 0) out_sel[b] = (float)(ns ? 0 : sel[b]);
}

// ---------------- host launcher ----------------
extern "C" void kernel_launch(void* const* d_in, const int* in_sizes, int n_in,
                              void* d_out, int out_size, void* d_ws, size_t ws_size,
                              hipStream_t stream) {
  const float* ae_in = (const float*)d_in[0];
  const float* uoh   = (const float*)d_in[1];
  const float* ent   = (const float*)d_in[2];
  const int*   enumv = (const int*)d_in[3];
  const unsigned char* selm = (const unsigned char*)d_in[4];
  const float* funcW = (const float*)d_in[5];
  const float* funcB = (const float*)d_in[6];
  const float* convW = (const float*)d_in[7];
  const float* convB = (const float*)d_in[8];
  const float* fc1W  = (const float*)d_in[9];
  const float* fc1B  = (const float*)d_in[10];
  const float* fc2W  = (const float*)d_in[11];
  const float* fc2B  = (const float*)d_in[12];
  const float* Wih   = (const float*)d_in[13];
  const float* Whh   = (const float*)d_in[14];
  const float* bih   = (const float*)d_in[15];
  const float* bhh   = (const float*)d_in[16];
  const float* projW = (const float*)d_in[17];
  const float* projB = (const float*)d_in[18];
  const float* newv  = (const float*)d_in[19];

  char* ws = (char*)d_ws;
  size_t off = 0;
  auto alloc = [&](size_t bytes) -> void* {
    void* p = ws + off;
    off = (off + bytes + 255) & ~(size_t)255;
    return p;
  };
  float*    keyt   = (float*)alloc((size_t)kB * kE * kK * 4);
  float*    tfe    = (float*)alloc((size_t)kB * kH * 4);
  float*    x      = (float*)alloc((size_t)kB * kH * 4);
  float*    keyavg = (float*)alloc((size_t)kB * kK * 4);
  unsigned* maskb  = (unsigned*)alloc((size_t)kB * 16 * 4);
  float*    h      = (float*)alloc((size_t)kB * kK * 4);
  float*    c      = (float*)alloc((size_t)kB * kK * 4);
  float*    ae     = (float*)alloc((size_t)kB * kAE * 4);
  bf16_t*   aeb    = (bf16_t*)alloc((size_t)kB * kAE * 2);
  unsigned* ended  = (unsigned*)alloc((size_t)kB * 4);
  int*      sel    = (int*)alloc((size_t)kB * 4);
  int*      gcnt   = (int*)alloc((size_t)kSEL * 4);
  bf16_t*   fc1b   = (bf16_t*)alloc((size_t)kH * kAE * 2);
  bf16_t*   fwb    = (bf16_t*)alloc((size_t)kH * kMUT * 2);
  bf16_t*   cwb    = (bf16_t*)alloc((size_t)kK * kD * 2);

  float* out_logits = (float*)d_out;                         // [B,64,E]
  float* out_units  = out_logits + (size_t)kB * kSEL * kE;   // [B,64,1]
  float* out_ae     = out_units + (size_t)kB * kSEL;         // [B,AE]
  float* out_sel    = out_ae + (size_t)kB * kAE;             // [B]

  k_convert<<<1024, 256, 0, stream>>>(fc1W, funcW, convW, fc1b, fwb, cwb);
  k_func_gemm<<<32, 256, 0, stream>>>(uoh, fwb, funcB, tfe);
  k_keyt_gemm<<<1024, 256, 0, stream>>>(ent, cwb, convB, keyt);
  k_batch_init<<<kB, 256, 0, stream>>>(keyt, enumv, newv, ae_in, keyavg, maskb,
                                       h, c, ae, aeb, ended, sel, gcnt);
  for (int i = 0; i < kSEL; ++i) {
    k_fc1_gemm<<<16, 256, 0, stream>>>(aeb, fc1b, fc1B, x);
    k_step<<<kB, 256, 0, stream>>>(i, x, tfe, fc2W, fc2B, Wih, Whh, bih, bhh,
                                   projW, projB, keyt, keyavg, enumv, selm,
                                   maskb, h, c, ae, aeb, ended, sel, gcnt,
                                   out_logits, out_units);
  }
  k_finalize<<<kB, 256, 0, stream>>>(ae, ae_in, selm, sel, out_ae, out_sel);
}
